// StructuralAttention_90838558311070
// MI455X (gfx1250) — matrix operand order repaired and checked by reading, not compiled, not run on hardware
//
#include <hip/hip_runtime.h>
#include <hip/hip_bf16.h>

typedef __attribute__((ext_vector_type(16))) _Float16 v16h;
typedef __attribute__((ext_vector_type(8)))  _Float16 h8;
typedef __attribute__((ext_vector_type(8)))  float    v8f;
typedef __attribute__((ext_vector_type(4)))  float    f4;

#define NODES 256
#define DIM   128
#define LDK   136   // padded f16 row stride for 128-wide rows (272B: bank-conflict free)
#define LDVT  264   // padded f16 row stride for 256-wide Vt rows (528B)
#define LDP   40    // per-wave staging stride (16x32 tile)

__device__ __forceinline__ h8 cvt8(const float* __restrict__ p) {
  f4 a = *(const f4*)p;
  f4 b = *(const f4*)(p + 4);
  h8 r;
  r[0]=(_Float16)a[0]; r[1]=(_Float16)a[1]; r[2]=(_Float16)a[2]; r[3]=(_Float16)a[3];
  r[4]=(_Float16)b[0]; r[5]=(_Float16)b[1]; r[6]=(_Float16)b[2]; r[7]=(_Float16)b[3];
  return r;
}

__device__ __forceinline__ v16h mk16(h8 lo, h8 hi) {
  v16h v;
#pragma unroll
  for (int i = 0; i < 8; ++i) { v[i] = lo[i]; v[i+8] = hi[i]; }
  return v;
}

__device__ __forceinline__ v8f wmma_f16(v16h a, v16h b, v8f c) {
  // D = A(16x32 f16) x B(32x16 f16) + C(16x16 f32)
  return __builtin_amdgcn_wmma_f32_16x16x32_f16(false, a, false, b, (short)0, c, false, false);
}

// Projection output tile: D = X_tile(16x128) * W^T(:, n*16..n*16+15) + b
__device__ __forceinline__ v8f proj_tile(const v16h xa[4],
                                         const float* __restrict__ W,
                                         const float* __restrict__ b,
                                         int n, int ln, int lh) {
  float bb = b[n * 16 + ln];
  v8f acc;
#pragma unroll
  for (int r = 0; r < 8; ++r) acc[r] = bb;
  const float* wrow = W + (size_t)(n * 16 + ln) * DIM;  // B[k][n'] = W[n'][k]: row of W
#pragma unroll
  for (int s = 0; s < 4; ++s) {
    h8 lo = cvt8(wrow + s * 32 + lh * 8);
    h8 hi = cvt8(wrow + s * 32 + lh * 8 + 16);
    acc = wmma_f16(xa[s], mk16(lo, hi), acc);
  }
  return acc;
}

__global__ __launch_bounds__(256)
void sattn_kernel(const float* __restrict__ x,
                  const float* __restrict__ Wq, const float* __restrict__ bq,
                  const float* __restrict__ Wk, const float* __restrict__ bk,
                  const float* __restrict__ Wv, const float* __restrict__ bv,
                  float* __restrict__ out) {
  __shared__ _Float16 sK[NODES * LDK];      //  69,632 B  K, row-major f16
  __shared__ _Float16 sVT[DIM * LDVT];      //  67,584 B  V^T, row-major f16
  __shared__ _Float16 sP[8][16 * LDP];      //  10,240 B  per-wave relayout staging
  // total 147,456 B -> 2 workgroups per 320KB WGP

  const int lane = threadIdx.x & 31;
  const int wv   = threadIdx.x >> 5;
  const int ln   = lane & 15;   // column (B/C/D) or row (A) within 16
  const int lh   = lane >> 4;   // half-select
  const size_t gbase = (size_t)blockIdx.x * NODES;

  // ---------------- Stage 1: K and V projections for this wave's 2 row-tiles --------
#pragma unroll
  for (int t = 0; t < 2; ++t) {
    const int mt    = 2 * wv + t;
    const int mbase = mt * 16;
    // A-operand: X rows (fp32 -> f16), kept for all 4 k-steps
    v16h xa[4];
    const float* xrow = x + (gbase + mbase + ln) * DIM;
#pragma unroll
    for (int s = 0; s < 4; ++s) {
      h8 lo = cvt8(xrow + s * 32 + lh * 8);
      h8 hi = cvt8(xrow + s * 32 + lh * 8 + 16);
      xa[s] = mk16(lo, hi);
    }
    // K -> sK (row-major)
#pragma unroll
    for (int n = 0; n < 8; ++n) {
      v8f acc = proj_tile(xa, Wk, bk, n, ln, lh);
#pragma unroll
      for (int r = 0; r < 8; ++r)
        sK[(mbase + r + 8 * lh) * LDK + n * 16 + ln] = (_Float16)acc[r];
    }
    // V -> sVT (transposed: VT[d][row])
#pragma unroll
    for (int n = 0; n < 8; ++n) {
      v8f acc = proj_tile(xa, Wv, bv, n, ln, lh);
#pragma unroll
      for (int r = 0; r < 8; ++r)
        sVT[(n * 16 + ln) * LDVT + mbase + r + 8 * lh] = (_Float16)acc[r];
    }
  }
  __syncthreads();

  const float scale = 0.08838834764831845f;  // 1/sqrt(128)

  // ---------------- Stage 2: flash attention for this wave's 2 Q row-tiles ----------
  for (int t = 0; t < 2; ++t) {
    const int rt    = 2 * wv + t;
    const int mbase = rt * 16;
    _Float16* pw = sP[wv];

    // Q projection -> A-layout registers, staged 32 columns at a time through sP
    v16h qa[4];
    {
      v16h xa[4];
      const float* xrow = x + (gbase + mbase + ln) * DIM;
#pragma unroll
      for (int s = 0; s < 4; ++s) {
        h8 lo = cvt8(xrow + s * 32 + lh * 8);
        h8 hi = cvt8(xrow + s * 32 + lh * 8 + 16);
        xa[s] = mk16(lo, hi);
      }
#pragma unroll
      for (int s = 0; s < 4; ++s) {   // k-chunk s covers Q cols s*32..s*32+31
#pragma unroll
        for (int u = 0; u < 2; ++u) {
          const int n = 2 * s + u;
          v8f acc = proj_tile(xa, Wq, bq, n, ln, lh);
#pragma unroll
          for (int r = 0; r < 8; ++r)
            pw[(r + 8 * lh) * LDP + u * 16 + ln] = (_Float16)acc[r];
        }
        asm volatile("s_wait_dscnt 0" ::: "memory");  // per-wave LDS RAW (C/D -> A)
        h8 lo = *(const h8*)&pw[ln * LDP + lh * 8];
        h8 hi = *(const h8*)&pw[ln * LDP + 16 + lh * 8];
        qa[s] = mk16(lo, hi);
        asm volatile("s_wait_dscnt 0" ::: "memory");  // WAR before next chunk's stores
      }
    }

    float m[8], l[8];
    v8f  o[8];
#pragma unroll
    for (int r = 0; r < 8; ++r) { m[r] = -1e30f; l[r] = 0.0f; }
#pragma unroll
    for (int n = 0; n < 8; ++n)
#pragma unroll
      for (int r = 0; r < 8; ++r) o[n][r] = 0.0f;

    for (int jb = 0; jb < 8; ++jb) {
      const int j0 = jb * 32;
      // S block [16 x 32] = Q_tile x K^T  (two 16x16 D tiles)
      v8f s0, s1;
#pragma unroll
      for (int r = 0; r < 8; ++r) { s0[r] = 0.0f; s1[r] = 0.0f; }
#pragma unroll
      for (int s = 0; s < 4; ++s) {
        const _Float16* k0 = &sK[(j0 + ln) * LDK + s * 32 + lh * 8];
        const _Float16* k1 = &sK[(j0 + 16 + ln) * LDK + s * 32 + lh * 8];
        s0 = wmma_f16(qa[s], mk16(*(const h8*)k0, *(const h8*)(k0 + 16)), s0);
        s1 = wmma_f16(qa[s], mk16(*(const h8*)k1, *(const h8*)(k1 + 16)), s1);
      }
      // online softmax update (row = r + 8*lh, reduce across 16-lane half)
      float f[8];
#pragma unroll
      for (int r = 0; r < 8; ++r) {
        float z0 = s0[r] * scale, z1 = s1[r] * scale;
        float mx = fmaxf(z0, z1);
#pragma unroll
        for (int d = 1; d < 16; d <<= 1) mx = fmaxf(mx, __shfl_xor(mx, d));
        float mn = fmaxf(m[r], mx);
        float fr = __expf(m[r] - mn);
        float p0 = __expf(z0 - mn);
        float p1 = __expf(z1 - mn);
        float sm = p0 + p1;
#pragma unroll
        for (int d = 1; d < 16; d <<= 1) sm += __shfl_xor(sm, d);
        l[r] = l[r] * fr + sm;
        m[r] = mn;
        f[r] = fr;
        s0[r] = p0; s1[r] = p1;
      }
#pragma unroll
      for (int n = 0; n < 8; ++n)
#pragma unroll
        for (int r = 0; r < 8; ++r) o[n][r] *= f[r];

      // P (C/D layout f32) -> per-wave LDS -> A-layout f16
#pragma unroll
      for (int r = 0; r < 8; ++r) {
        const int M = r + 8 * lh;
        pw[M * LDP + ln]      = (_Float16)s0[r];
        pw[M * LDP + 16 + ln] = (_Float16)s1[r];
      }
      asm volatile("s_wait_dscnt 0" ::: "memory");
      h8 plo = *(const h8*)&pw[ln * LDP + lh * 8];
      h8 phi = *(const h8*)&pw[ln * LDP + 16 + lh * 8];
      v16h pa = mk16(plo, phi);

      // O += P_block(16x32) x V_block(32x128)  (B from V^T rows, contiguous in j)
#pragma unroll
      for (int n = 0; n < 8; ++n) {
        const _Float16* vt = &sVT[(n * 16 + ln) * LDVT + j0 + lh * 8];
        o[n] = wmma_f16(pa, mk16(*(const h8*)vt, *(const h8*)(vt + 16)), o[n]);
      }
    }

    // normalize and write out (fp32)
    float inv[8];
#pragma unroll
    for (int r = 0; r < 8; ++r) inv[r] = 1.0f / l[r];
#pragma unroll
    for (int n = 0; n < 8; ++n)
#pragma unroll
      for (int r = 0; r < 8; ++r)
        out[(gbase + mbase + r + 8 * lh) * DIM + n * 16 + ln] = o[n][r] * inv[r];
  }
}

extern "C" void kernel_launch(void* const* d_in, const int* in_sizes, int n_in,
                              void* d_out, int out_size, void* d_ws, size_t ws_size,
                              hipStream_t stream) {
  const float* x  = (const float*)d_in[0];
  // d_in[1] = batch (int64): contiguous equal segments -> implicit in blockIdx
  const float* Wq = (const float*)d_in[2];
  const float* bq = (const float*)d_in[3];
  const float* Wk = (const float*)d_in[4];
  const float* bk = (const float*)d_in[5];
  const float* Wv = (const float*)d_in[6];
  const float* bv = (const float*)d_in[7];
  float* out = (float*)d_out;

  const int nGraphs = in_sizes[0] / (NODES * DIM);  // 512
  sattn_kernel<<<dim3(nGraphs), dim3(256), 0, stream>>>(x, Wq, bq, Wk, bk, Wv, bv, out);
}
